// MultiChannelMultiAttention_18494129176680
// MI455X (gfx1250) — compile-verified, matrix operand order checked by reading, not run on hardware
//
#include <hip/hip_runtime.h>
#include <math.h>

typedef __attribute__((ext_vector_type(16))) _Float16 v16h;
typedef __attribute__((ext_vector_type(8)))  _Float16 v8h;
typedef __attribute__((ext_vector_type(8)))  float    v8f;

// ---------------------------------------------------------------------------
// Load one lane's 16-element f16 fragment: two contiguous 8xf16 (b128) runs at
// k = base and k = base+16 (CDNA5 ISA 7.12.2 16-bit A/B layout with the lane's
// half-select already folded into `p`). `p` must be 16B aligned.
// ---------------------------------------------------------------------------
__device__ __forceinline__ v16h load_frag(const _Float16* __restrict__ p)
{
  const v8h lo = *(const v8h*)(p);
  const v8h hi = *(const v8h*)(p + 16);
  v16h f;
#pragma unroll
  for (int i = 0; i < 8; ++i) { f[i] = lo[i]; f[8 + i] = hi[i]; }
  return f;
}

// ---------------------------------------------------------------------------
// 16x16-tile GEMM, all-f16 operands, f32 accumulate via v_wmma_f32_16x16x32_f16.
// C[m,n] = sum_k A[m,k] * Bt[n,k]  (+ bias1[n] + bias2[n])
// A: (M,K) row-major f16, lda;  Bt: (N,K) row-major f16, ldb (K contiguous).
// Optional f32 and/or f16 outputs. One wave per tile; grid=(N/16, M/16, batch).
// ---------------------------------------------------------------------------
__global__ void gemm_f16_kernel(
    const _Float16* __restrict__ A,  long long sAb, int lda,
    const _Float16* __restrict__ Bt, long long sBb, int ldb,
    float* __restrict__ C,   long long sCb, int ldc,
    _Float16* __restrict__ C16, long long sC16b,
    const float* __restrict__ bias1, const float* __restrict__ bias2,
    int K)
{
  const int tn = blockIdx.x;
  const int tm = blockIdx.y;
  const int bz = blockIdx.z;

  const int lane    = threadIdx.x & 31;
  const int halfsel = lane >> 4;     // 0: lanes 0-15, 1: lanes 16-31
  const int mr      = lane & 15;
  const int arow    = tm * 16 + mr;  // A row for this lane
  const int ncol    = tn * 16 + mr;  // B/C column for this lane

  const _Float16* pa = A  + (size_t)bz * sAb + (size_t)arow * lda + halfsel * 8;
  const _Float16* pb = Bt + (size_t)bz * sBb + (size_t)ncol * ldb + halfsel * 8;

  v8f acc = {};
  for (int kk = 0; kk < K; kk += 32) {
    if (kk + 32 < K) {               // gfx1250 global_prefetch_b8 path
      __builtin_prefetch(pa + kk + 32, 0, 3);
      __builtin_prefetch(pb + kk + 32, 0, 3);
    }
    const v16h af = load_frag(pa + kk);
    const v16h bf = load_frag(pb + kk);
    acc = __builtin_amdgcn_wmma_f32_16x16x32_f16(
        /*neg_a=*/false, af, /*neg_b=*/false, bf,
        /*c_mod=*/(short)0, acc, /*reuse_a=*/false, /*reuse_b=*/false);
  }

  float badd = 0.0f;
  if (bias1) badd += bias1[ncol];
  if (bias2) badd += bias2[ncol];

#pragma unroll
  for (int r = 0; r < 8; ++r) {
    const int m = tm * 16 + r + halfsel * 8;   // C/D layout: vgpr r -> row r (+8)
    const float val = acc[r] + badd;
    if (C)   (C   + (size_t)bz * sCb  )[(size_t)m * ldc + ncol] = val;
    if (C16) (C16 + (size_t)bz * sC16b)[(size_t)m * ldc + ncol] = (_Float16)val;
  }
}

// --------------------------- f32 -> f16 convert ----------------------------
__global__ void cvt_f16_kernel(const float* __restrict__ src,
                               _Float16* __restrict__ dst, int n)
{
  const int i = blockIdx.x * blockDim.x + threadIdx.x;
  if (i < n) dst[i] = (_Float16)src[i];
}

// ----------------- transpose (B,L,D)f32 -> (B,D,L)f16 ----------------------
__global__ void transpose_cvt_kernel(const float* __restrict__ src,
                                     _Float16* __restrict__ dst)
{
  const int d = blockIdx.x;  // 128
  const int b = blockIdx.y;  // 4
  const float* s = src + (size_t)b * 512 * 128;
  _Float16*    o = dst + (size_t)b * 128 * 512 + (size_t)d * 512;
  for (int j = threadIdx.x; j < 512; j += blockDim.x)
    o[j] = (_Float16)s[(size_t)j * 128 + d];
}

// ---------------------------------------------------------------------------
// Additive attention scores: s3[b,i,j] = sum_d tanh(x[b,i,d] + x[b,j,d]),
// only j <= i needed (softmax zeroes the rest). grid = (L, B), 512 threads.
// ---------------------------------------------------------------------------
__global__ void s3_kernel(const float* __restrict__ x, float* __restrict__ s3)
{
  const int i = blockIdx.x, b = blockIdx.y;
  __shared__ float xi[128];
  const float* xb = x + (size_t)b * 512 * 128;
  for (int d = threadIdx.x; d < 128; d += blockDim.x) xi[d] = xb[(size_t)i * 128 + d];
  __syncthreads();
  for (int j = threadIdx.x; j <= i; j += blockDim.x) {
    const float* xj = xb + (size_t)j * 128;
    float s = 0.0f;
#pragma unroll 4
    for (int d = 0; d < 128; ++d) s += tanhf(xi[d] + xj[d]);
    s3[(((size_t)b * 512) + i) * 512 + j] = s;
  }
}

// ---------------------------------------------------------------------------
// Causal softmax; reads f32 scores, writes f16 probabilities (masked -> 0).
// scale = scale_const * (*scale_ptr). grid = (L, B), 256 threads.
// ---------------------------------------------------------------------------
__global__ void softmax_kernel(const float* __restrict__ s,
                               _Float16* __restrict__ p,
                               const float* __restrict__ scale_ptr,
                               float scale_const)
{
  const int i = blockIdx.x, b = blockIdx.y;
  const float scale = scale_const * (scale_ptr ? scale_ptr[0] : 1.0f);
  const float* row = s + (((size_t)b * 512) + i) * 512;
  _Float16*    po  = p + (((size_t)b * 512) + i) * 512;
  __shared__ float red[256];
  const int tid = threadIdx.x;
  const int n = i + 1;

  float mx = -3.4e38f;
  for (int j = tid; j < n; j += 256) mx = fmaxf(mx, row[j] * scale);
  red[tid] = mx; __syncthreads();
  for (int o = 128; o > 0; o >>= 1) { if (tid < o) red[tid] = fmaxf(red[tid], red[tid + o]); __syncthreads(); }
  mx = red[0]; __syncthreads();

  float sum = 0.0f;
  for (int j = tid; j < n; j += 256) sum += __expf(row[j] * scale - mx);
  red[tid] = sum; __syncthreads();
  for (int o = 128; o > 0; o >>= 1) { if (tid < o) red[tid] += red[tid + o]; __syncthreads(); }
  const float inv = 1.0f / red[0];
  __syncthreads();

  for (int j = tid; j < 512; j += 256)
    po[j] = (j < n) ? (_Float16)(__expf(row[j] * scale - mx) * inv) : (_Float16)0.0f;
}

__global__ void attn_combine_kernel(const float* __restrict__ a1,
                                    const float* __restrict__ a2,
                                    const float* __restrict__ a3,
                                    const float* __restrict__ attn_w,
                                    float* __restrict__ attn_out)
{
  const float w0 = attn_w[0], w1 = attn_w[1], w2 = attn_w[2];
  const float inv = 1.0f / (w0 + w1 + w2);
  const size_t idx = (size_t)blockIdx.x * blockDim.x + threadIdx.x;
  attn_out[idx] = (a1[idx] * w0 + a2[idx] * w1 + a3[idx] * w2) * inv;
}

// ---------------------------------------------------------------------------
// BatchNorm over (B,L) per channel (biased variance), f16 output (feeds GEMM).
// grid = 128 channels. in = in1 (+ in2 if non-null).
// ---------------------------------------------------------------------------
__global__ void bn_kernel(const float* __restrict__ in1,
                          const float* __restrict__ in2,
                          const float* __restrict__ g,
                          const float* __restrict__ bb,
                          _Float16* __restrict__ out)
{
  const int c = blockIdx.x;
  __shared__ float rs[256], rs2[256];
  float s = 0.0f, s2 = 0.0f;
  for (int r = threadIdx.x; r < 2048; r += 256) {
    float v = in1[(size_t)r * 128 + c];
    if (in2) v += in2[(size_t)r * 128 + c];
    s += v; s2 += v * v;
  }
  rs[threadIdx.x] = s; rs2[threadIdx.x] = s2; __syncthreads();
  for (int o = 128; o > 0; o >>= 1) {
    if (threadIdx.x < o) { rs[threadIdx.x] += rs[threadIdx.x + o]; rs2[threadIdx.x] += rs2[threadIdx.x + o]; }
    __syncthreads();
  }
  const float mean  = rs[0] * (1.0f / 2048.0f);
  const float var   = rs2[0] * (1.0f / 2048.0f) - mean * mean;
  const float scale = g[c] * rsqrtf(var + 1e-5f);
  const float shift = bb[c] - mean * scale;
  for (int r = threadIdx.x; r < 2048; r += 256) {
    float v = in1[(size_t)r * 128 + c];
    if (in2) v += in2[(size_t)r * 128 + c];
    out[(size_t)r * 128 + c] = (_Float16)(v * scale + shift);
  }
}

// ---------------------------------------------------------------------------
// LSTM recurrence, one layer of 3 branches (blockIdx.x = branch). Gate order
// i,f,g,o. Gate pre-activations G (B,L,4H) precomputed by WMMA GEMM; only
// z += h_{t-1} @ Whh^T here. h/c/z in LDS. Writes f16 and/or f32 h sequence.
// ---------------------------------------------------------------------------
__global__ void __launch_bounds__(1024) lstm_rec_kernel(
    const float* __restrict__ G0, const float* __restrict__ G1, const float* __restrict__ G2,
    const float* __restrict__ W0, const float* __restrict__ W1, const float* __restrict__ W2,
    _Float16* __restrict__ H0, _Float16* __restrict__ H1, _Float16* __restrict__ H2,
    float* __restrict__ O0, float* __restrict__ O1, float* __restrict__ O2)
{
  const int br = blockIdx.x;
  const float* G  = (br == 0) ? G0 : ((br == 1) ? G1 : G2);
  const float* Wh = (br == 0) ? W0 : ((br == 1) ? W1 : W2);
  _Float16*    H  = (br == 0) ? H0 : ((br == 1) ? H1 : H2);
  float*       O  = (br == 0) ? O0 : ((br == 1) ? O1 : O2);

  __shared__ float z[4][512];
  __shared__ float hs[4][128];
  __shared__ float cs[4][128];
  const int tid = threadIdx.x;
  if (tid < 512) { hs[tid >> 7][tid & 127] = 0.0f; cs[tid >> 7][tid & 127] = 0.0f; }
  __syncthreads();

  for (int t = 0; t < 512; ++t) {
    for (int idx = tid; idx < 2048; idx += 1024) {
      const int b = idx >> 9, n = idx & 511;
      const float* wr = Wh + (size_t)n * 128;
      const float* hb = hs[b];
      float acc = G[(((size_t)b * 512) + t) * 512 + n];
#pragma unroll 8
      for (int j = 0; j < 128; ++j) acc += hb[j] * wr[j];
      z[b][n] = acc;
    }
    __syncthreads();
    if (tid < 512) {
      const int b = tid >> 7, j = tid & 127;
      const float zi = z[b][j], zf = z[b][128 + j], zg = z[b][256 + j], zo = z[b][384 + j];
      const float ig = 1.0f / (1.0f + __expf(-zi));
      const float fg = 1.0f / (1.0f + __expf(-zf));
      const float gg = tanhf(zg);
      const float og = 1.0f / (1.0f + __expf(-zo));
      const float c  = fg * cs[b][j] + ig * gg;
      const float h  = og * tanhf(c);
      cs[b][j] = c; hs[b][j] = h;
      const size_t off = (((size_t)b * 512) + t) * 128 + j;
      if (H) H[off] = (_Float16)h;
      if (O) O[off] = h;
    }
    __syncthreads();
  }
}

// ---------------------------------------------------------------------------
// Weighted sum of branch outputs + LayerNorm over last dim.
// grid = 2048 rows, 128 threads.
// ---------------------------------------------------------------------------
__global__ void final_ln_kernel(const float* __restrict__ l1,
                                const float* __restrict__ l2,
                                const float* __restrict__ l3,
                                const float* __restrict__ sum_w,
                                const float* __restrict__ g,
                                const float* __restrict__ bb,
                                float* __restrict__ out)
{
  const int r = blockIdx.x;
  const int c = threadIdx.x;
  __shared__ float red[128];
  const float w0 = sum_w[0], w1 = sum_w[1], w2 = sum_w[2];
  const float inv = 1.0f / (w0 + w1 + w2);
  const size_t off = (size_t)r * 128 + c;
  const float v = (l1[off] * w0 + l2[off] * w1 + l3[off] * w2) * inv;

  red[c] = v; __syncthreads();
  for (int o = 64; o > 0; o >>= 1) { if (c < o) red[c] += red[c + o]; __syncthreads(); }
  const float mean = red[0] * (1.0f / 128.0f);
  __syncthreads();
  const float d = v - mean;
  red[c] = d * d; __syncthreads();
  for (int o = 64; o > 0; o >>= 1) { if (c < o) red[c] += red[c + o]; __syncthreads(); }
  const float var = red[0] * (1.0f / 128.0f);
  out[off] = d * rsqrtf(var + 1e-5f) * g[c] + bb[c];
}

// ---------------------------------------------------------------------------
extern "C" void kernel_launch(void* const* d_in, const int* in_sizes, int n_in,
                              void* d_out, int out_size, void* d_ws, size_t ws_size,
                              hipStream_t stream)
{
  (void)in_sizes; (void)n_in; (void)out_size; (void)ws_size;
  const float* x          = (const float*)d_in[0];
  const float* attn_w     = (const float*)d_in[1];
  const float* attn_scale = (const float*)d_in[2];
  const float* Wq         = (const float*)d_in[3];
  const float* Wk         = (const float*)d_in[4];
  const float* bk         = (const float*)d_in[5];
  const float* Wv         = (const float*)d_in[6];
  const float* bn_g[3] = {(const float*)d_in[7],  (const float*)d_in[9],  (const float*)d_in[11]};
  const float* bn_b[3] = {(const float*)d_in[8],  (const float*)d_in[10], (const float*)d_in[12]};
  const float* Wih[3]  = {(const float*)d_in[13], (const float*)d_in[17], (const float*)d_in[21]};
  const float* Whh[3]  = {(const float*)d_in[14], (const float*)d_in[18], (const float*)d_in[22]};
  const float* bih[3]  = {(const float*)d_in[15], (const float*)d_in[19], (const float*)d_in[23]};
  const float* bhh[3]  = {(const float*)d_in[16], (const float*)d_in[20], (const float*)d_in[24]};
  const float* ln_g  = (const float*)d_in[25];
  const float* ln_b  = (const float*)d_in[26];
  const float* sum_w = (const float*)d_in[27];
  float* out = (float*)d_out;

  // ---- workspace carve (256B-aligned chunks) ----
  char*  base = (char*)d_ws;
  size_t off  = 0;
  auto alloc = [&](size_t bytes) -> char* {
    char* p = base + off;
    off += (bytes + 255) & ~(size_t)255;
    return p;
  };
  const size_t RD = 2048ull * 128;        // (B*L, D) elements
  const size_t SS = 4ull * 512 * 512;     // (B, L, L) elements
  const size_t GG = 2048ull * 512;        // (B*L, 4H) elements

  _Float16* xh   = (_Float16*)alloc(RD * 2);
  _Float16* qh   = (_Float16*)alloc(RD * 2);
  _Float16* kh   = (_Float16*)alloc(RD * 2);
  _Float16* xTh  = (_Float16*)alloc(RD * 2);      // (B, D, L)
  _Float16* vTh  = (_Float16*)alloc(RD * 2);      // (B, D, L)
  _Float16* Wqh  = (_Float16*)alloc(128 * 128 * 2);
  _Float16* Wkh  = (_Float16*)alloc(128 * 128 * 2);
  _Float16* Wvh  = (_Float16*)alloc(128 * 128 * 2);
  _Float16* Wihh[3];
  for (int i = 0; i < 3; ++i) Wihh[i] = (_Float16*)alloc(2ull * 512 * 128 * 2);
  _Float16* ph1  = (_Float16*)alloc(SS * 2);
  _Float16* ph2  = (_Float16*)alloc(SS * 2);
  _Float16* ph3  = (_Float16*)alloc(SS * 2);
  _Float16* boh1 = (_Float16*)alloc(RD * 2);
  _Float16* boh2 = (_Float16*)alloc(RD * 2);
  _Float16* boh3 = (_Float16*)alloc(RD * 2);
  _Float16* h1h1 = (_Float16*)alloc(RD * 2);
  _Float16* h1h2 = (_Float16*)alloc(RD * 2);
  _Float16* h1h3 = (_Float16*)alloc(RD * 2);
  float* vf  = (float*)alloc(RD * 4);
  float* s1  = (float*)alloc(SS * 4);
  float* s2  = (float*)alloc(SS * 4);
  float* s3  = (float*)alloc(SS * 4);
  float* a1  = (float*)alloc(RD * 4);
  float* a2  = (float*)alloc(RD * 4);
  float* a3  = (float*)alloc(RD * 4);
  float* ao  = (float*)alloc(RD * 4);
  float* G1  = (float*)alloc(GG * 4);
  float* G2  = (float*)alloc(GG * 4);
  float* G3  = (float*)alloc(GG * 4);
  float* li1 = (float*)alloc(RD * 4);
  float* li2 = (float*)alloc(RD * 4);
  float* li3 = (float*)alloc(RD * 4);

  const long long sX = 512 * 128;    // per-batch (L,D) / (D,L) stride
  const long long sS = 512 * 512;    // per-batch (L,L) stride
  const long long WL = 512 * 128;    // per-layer Wih/Whh stride

  // ---- one-time f16 conversions / transposes ----
  cvt_f16_kernel<<<dim3((RD + 255) / 256), 256, 0, stream>>>(x, xh, (int)RD);
  cvt_f16_kernel<<<dim3(64), 256, 0, stream>>>(Wq, Wqh, 128 * 128);
  cvt_f16_kernel<<<dim3(64), 256, 0, stream>>>(Wk, Wkh, 128 * 128);
  cvt_f16_kernel<<<dim3(64), 256, 0, stream>>>(Wv, Wvh, 128 * 128);
  for (int i = 0; i < 3; ++i)
    cvt_f16_kernel<<<dim3(512), 256, 0, stream>>>(Wih[i], Wihh[i], 2 * 512 * 128);
  transpose_cvt_kernel<<<dim3(128, 4), 256, 0, stream>>>(x, xTh);

  // ---- projections: q,k straight to f16; v to f32 (transposed next) ----
  gemm_f16_kernel<<<dim3(8, 128, 1), 32, 0, stream>>>(xh, 0, 128, Wqh, 0, 128, nullptr, 0, 128, qh, 0, nullptr, nullptr, 128);
  gemm_f16_kernel<<<dim3(8, 128, 1), 32, 0, stream>>>(xh, 0, 128, Wkh, 0, 128, nullptr, 0, 128, kh, 0, bk, nullptr, 128);
  gemm_f16_kernel<<<dim3(8, 128, 1), 32, 0, stream>>>(xh, 0, 128, Wvh, 0, 128, vf, 0, 128, nullptr, 0, nullptr, nullptr, 128);
  transpose_cvt_kernel<<<dim3(128, 4), 256, 0, stream>>>(vf, vTh);

  // ---- scores ----
  gemm_f16_kernel<<<dim3(32, 32, 4), 32, 0, stream>>>(xh, sX, 128, xh, sX, 128, s1, sS, 512, nullptr, 0, nullptr, nullptr, 128);
  gemm_f16_kernel<<<dim3(32, 32, 4), 32, 0, stream>>>(qh, sX, 128, kh, sX, 128, s2, sS, 512, nullptr, 0, nullptr, nullptr, 128);
  s3_kernel<<<dim3(512, 4), 512, 0, stream>>>(x, s3);

  softmax_kernel<<<dim3(512, 4), 256, 0, stream>>>(s1, ph1, nullptr, 1.0f);
  softmax_kernel<<<dim3(512, 4), 256, 0, stream>>>(s2, ph2, attn_scale, 0.08838834764831845f); // 1/sqrt(128)
  softmax_kernel<<<dim3(512, 4), 256, 0, stream>>>(s3, ph3, nullptr, 1.0f);

  // ---- P@V (A = f16 probs, B = transposed f16 values, K=512) ----
  gemm_f16_kernel<<<dim3(8, 32, 4), 32, 0, stream>>>(ph1, sS, 512, xTh, sX, 512, a1, sX, 128, nullptr, 0, nullptr, nullptr, 512);
  gemm_f16_kernel<<<dim3(8, 32, 4), 32, 0, stream>>>(ph2, sS, 512, vTh, sX, 512, a2, sX, 128, nullptr, 0, nullptr, nullptr, 512);
  gemm_f16_kernel<<<dim3(8, 32, 4), 32, 0, stream>>>(ph3, sS, 512, xTh, sX, 512, a3, sX, 128, nullptr, 0, nullptr, nullptr, 512);
  attn_combine_kernel<<<dim3(1024), 256, 0, stream>>>(a1, a2, a3, attn_w, ao);

  // ---- BatchNorms (write f16, feed gate GEMMs) ----
  bn_kernel<<<dim3(128), 256, 0, stream>>>(x, nullptr, bn_g[0], bn_b[0], boh1);
  bn_kernel<<<dim3(128), 256, 0, stream>>>(x, ao,      bn_g[1], bn_b[1], boh2);
  bn_kernel<<<dim3(128), 256, 0, stream>>>(ao, nullptr, bn_g[2], bn_b[2], boh3);

  // ---- LSTM layer 0: gate GEMMs + recurrence (h seq -> f16) ----
  gemm_f16_kernel<<<dim3(32, 128, 1), 32, 0, stream>>>(boh1, 0, 128, Wihh[0], 0, 128, G1, 0, 512, nullptr, 0, bih[0], bhh[0], 128);
  gemm_f16_kernel<<<dim3(32, 128, 1), 32, 0, stream>>>(boh2, 0, 128, Wihh[1], 0, 128, G2, 0, 512, nullptr, 0, bih[1], bhh[1], 128);
  gemm_f16_kernel<<<dim3(32, 128, 1), 32, 0, stream>>>(boh3, 0, 128, Wihh[2], 0, 128, G3, 0, 512, nullptr, 0, bih[2], bhh[2], 128);
  lstm_rec_kernel<<<dim3(3), 1024, 0, stream>>>(G1, G2, G3, Whh[0], Whh[1], Whh[2],
                                                h1h1, h1h2, h1h3, nullptr, nullptr, nullptr);

  // ---- LSTM layer 1: gate GEMMs + recurrence (h seq -> f32 line outputs) ----
  gemm_f16_kernel<<<dim3(32, 128, 1), 32, 0, stream>>>(h1h1, 0, 128, Wihh[0] + WL, 0, 128, G1, 0, 512, nullptr, 0, bih[0] + 512, bhh[0] + 512, 128);
  gemm_f16_kernel<<<dim3(32, 128, 1), 32, 0, stream>>>(h1h2, 0, 128, Wihh[1] + WL, 0, 128, G2, 0, 512, nullptr, 0, bih[1] + 512, bhh[1] + 512, 128);
  gemm_f16_kernel<<<dim3(32, 128, 1), 32, 0, stream>>>(h1h3, 0, 128, Wihh[2] + WL, 0, 128, G3, 0, 512, nullptr, 0, bih[2] + 512, bhh[2] + 512, 128);
  lstm_rec_kernel<<<dim3(3), 1024, 0, stream>>>(G1, G2, G3, Whh[0] + WL, Whh[1] + WL, Whh[2] + WL,
                                                nullptr, nullptr, nullptr, li1, li2, li3);

  // ---- weighted sum + LayerNorm ----
  final_ln_kernel<<<dim3(2048), 128, 0, stream>>>(li1, li2, li3, sum_w, ln_g, ln_b, out);
}
